// PMHCProjectionHead_4080218931898
// MI455X (gfx1250) — matrix operand order repaired and checked by reading, not compiled, not run on hardware
//
#include <hip/hip_runtime.h>

typedef __attribute__((ext_vector_type(16))) __bf16 v16bf;
typedef __attribute__((ext_vector_type(8)))  float  v8f;

union BF2 { __bf16 h[2]; unsigned u; };
__device__ __forceinline__ unsigned pkbf(float a, float b) {
  BF2 t; t.h[0] = (__bf16)a; t.h[1] = (__bf16)b; return t.u;
}

union VB { v16bf v; unsigned d[8]; uint4 q[2]; };

// Build a 16x32 bf16 A-fragment slice from a f32 row pointer.
// Per ISA A-layout (16-bit 16x32): this lane's 16 elements are
// cols [base .. base+7] and [base+16 .. base+23] of its row, where the
// caller passes p = row + 32*f + 8*half.
__device__ __forceinline__ v16bf ld_a_f32(const float* p) {
  const float4 a = *(const float4*)(p);
  const float4 b = *(const float4*)(p + 4);
  const float4 c = *(const float4*)(p + 16);
  const float4 e = *(const float4*)(p + 20);
  v16bf r;
  r[0]=(__bf16)a.x;  r[1]=(__bf16)a.y;  r[2]=(__bf16)a.z;  r[3]=(__bf16)a.w;
  r[4]=(__bf16)b.x;  r[5]=(__bf16)b.y;  r[6]=(__bf16)b.z;  r[7]=(__bf16)b.w;
  r[8]=(__bf16)c.x;  r[9]=(__bf16)c.y;  r[10]=(__bf16)c.z; r[11]=(__bf16)c.w;
  r[12]=(__bf16)e.x; r[13]=(__bf16)e.y; r[14]=(__bf16)e.z; r[15]=(__bf16)e.w;
  return r;
}

// ---------------------------------------------------------------------------
// Stage 1: fused  U[b] = Ac^T @ ((emb[b]*mask[b]) @ Bc)   ->  bf16 [B, 256]
// One wave per batch sample (grid-stride). Streams emb exactly once.
// Mask is folded into the Y tiles (row scaling commutes with X @ Bc):
// 16 muls per chunk instead of 128-wide X masking.
// ---------------------------------------------------------------------------
template<int L, int NCHUNK>
__global__ void encoder_kernel(const float* __restrict__ emb, const float* __restrict__ msk,
                               const float* __restrict__ Bc,  const float* __restrict__ Ac,
                               __bf16* __restrict__ Uout, int Bn)
{
  const int lane = threadIdx.x & 31;
  const int wv   = blockIdx.x * (blockDim.x >> 5) + (threadIdx.x >> 5);
  const int nwv  = gridDim.x * (blockDim.x >> 5);
  const int m  = lane & 15;
  const int hh = lane >> 4;

  // Bc [128,16] as 4 B-fragments (K=32 each), bf16, register resident.
  // B-layout: lane n=m, element j covers k = 32f + (j<8 ? 8*hh + j : 16 + 8*hh + j-8).
  v16bf bcf[4];
  #pragma unroll
  for (int f = 0; f < 4; ++f) {
    #pragma unroll
    for (int j = 0; j < 16; ++j) {
      int k = 32*f + ((j < 8) ? (8*hh + j) : (16 + 8*hh + (j - 8)));
      bcf[f][j] = (__bf16)Bc[k*16 + m];
    }
  }

  // Ac^T per 32-row chunk as A-fragments (M=q=16, K=l=32), register resident.
  v16bf acf[NCHUNK];
  #pragma unroll
  for (int cch = 0; cch < NCHUNK; ++cch) {
    #pragma unroll
    for (int j = 0; j < 16; ++j) {
      int l = cch*32 + 16*hh + j;
      acf[cch][j] = (__bf16)Ac[l*16 + m];   // Ac[l][q=m]
    }
  }

  for (int b = wv; b < Bn; b += nwv) {
    const float* xb = emb + (size_t)b * L * 128;
    const float* mb = msk + (size_t)b * L;
    v8f u = {0,0,0,0,0,0,0,0};

    #pragma unroll
    for (int cch = 0; cch < NCHUNK; ++cch) {
      const int r0 = cch*32 + m;        // rows of tile 0 (this lane's A row)
      const int r1 = r0 + 16;           // rows of tile 1
      const float* p0 = xb + (size_t)r0 * 128;
      const float* p1 = xb + (size_t)r1 * 128;
      if (cch + 1 < NCHUNK) {           // prefetch next chunk of the stream
        __builtin_prefetch(p0 + 32*128, 0, 0);
        __builtin_prefetch(p1 + 32*128, 0, 0);
      }

      v8f y0 = {0,0,0,0,0,0,0,0};
      v8f y1 = {0,0,0,0,0,0,0,0};
      #pragma unroll
      for (int f = 0; f < 4; ++f) {     // K = 128 over Bc
        v16bf a0 = ld_a_f32(p0 + 32*f + 8*hh);
        v16bf a1 = ld_a_f32(p1 + 32*f + 8*hh);
        y0 = __builtin_amdgcn_wmma_f32_16x16x32_bf16(false, a0, false, bcf[f], (short)0, y0, false, false);
        y1 = __builtin_amdgcn_wmma_f32_16x16x32_bf16(false, a1, false, bcf[f], (short)0, y1, false, false);
      }

      // Fold the mask in on the 16x16 Y tiles: C-layout VGPR i holds row
      // 8*hh + i, so tile0 needs mask[c*32 + 8*hh + i], tile1 the +16 rows.
      {
        const float* mr = mb + cch*32 + 8*hh;
        const float4 ma = *(const float4*)(mr);
        const float4 mv = *(const float4*)(mr + 4);
        const float4 mc = *(const float4*)(mr + 16);
        const float4 md = *(const float4*)(mr + 20);
        y0[0]*=ma.x; y0[1]*=ma.y; y0[2]*=ma.z; y0[3]*=ma.w;
        y0[4]*=mv.x; y0[5]*=mv.y; y0[6]*=mv.z; y0[7]*=mv.w;
        y1[0]*=mc.x; y1[1]*=mc.y; y1[2]*=mc.z; y1[3]*=mc.w;
        y1[4]*=md.x; y1[5]*=md.y; y1[6]*=md.z; y1[7]*=md.w;
      }

      // Repack Y0/Y1 (f32 C-layout) into one 32x16 bf16 B-fragment.
      // Low lane n needs Y0 rows 0..15 col n; high lane n needs Y1 rows 0..15.
      // Own regs give half; the other half comes from the partner lane (xor 16).
      unsigned pw0[4], pw1[4];
      #pragma unroll
      for (int i = 0; i < 4; ++i) {
        pw0[i] = pkbf(y0[2*i], y0[2*i+1]);
        pw1[i] = pkbf(y1[2*i], y1[2*i+1]);
      }
      VB yb;
      #pragma unroll
      for (int i = 0; i < 4; ++i) {
        unsigned snd = (hh == 0) ? pw1[i] : pw0[i];
        unsigned rcv = __shfl_xor(snd, 16, 32);
        yb.d[i]     = (hh == 0) ? pw0[i] : rcv;
        yb.d[4 + i] = (hh == 0) ? rcv    : pw1[i];
      }

      u = __builtin_amdgcn_wmma_f32_16x16x32_bf16(false, acf[cch], false, yb.v, (short)0, u, false, false);
    }

    // U[b] 16x16, flatten q*16+r; C-layout row = 8*hh + i, col = m.
    __bf16* up = Uout + (size_t)b * 256;
    #pragma unroll
    for (int i = 0; i < 8; ++i)
      up[(8*hh + i)*16 + m] = (__bf16)u[i];
  }
}

// ---------------------------------------------------------------------------
// Weight packer: f32 [K,N] row-major -> zero-padded bf16 32x16 B-fragment
// blocks, layout dword index = ((nt*KC + kc)*32 + lane)*8 + i.
// ---------------------------------------------------------------------------
__global__ void pack_w_kernel(const float* __restrict__ W, unsigned* __restrict__ blk,
                              int K, int N, int KC, int NT)
{
  int idx = blockIdx.x * blockDim.x + threadIdx.x;
  int total = KC * NT * 256;
  if (idx >= total) return;
  int i     = idx & 7;
  int lane  = (idx >> 3) & 31;
  int blkid = idx >> 8;
  int kc = blkid % KC;
  int nt = blkid / KC;
  int n  = nt*16 + (lane & 15);
  int k0 = 32*kc + 8*(lane >> 4) + 2*i + ((i >= 4) ? 8 : 0);
  float x0 = (k0   < K && n < N) ? W[(size_t)k0     * N + n] : 0.f;
  float x1 = (k0+1 < K && n < N) ? W[(size_t)(k0+1) * N + n] : 0.f;
  blk[idx] = pkbf(x0, x1);
}

// ---------------------------------------------------------------------------
// Stage 2: C = A[bf16 B x K] @ Wblk (+bias) with fused epilogue.
// One wave owns 16 rows x full feature dim (NT tiles) so LayerNorm reduces
// in-register via shfl butterflies (xor 1/2/4/8 stays within 16-lane halves,
// matching the C layout: rows 0..7 in lanes 0..15, rows 8..15 in 16..31).
// MODE 0: store bf16 (no bias). MODE 1: +bias, LN, ReLU -> bf16.
// MODE 2: +bias -> f32 strided store into concatenated output.
// ---------------------------------------------------------------------------
template<int KC, int NT, int MODE>
__global__ void gemm_kernel(const __bf16* __restrict__ A, const unsigned* __restrict__ Wblk,
                            const float* __restrict__ bias, const float* __restrict__ gam,
                            const float* __restrict__ beta, int d, int Bn,
                            __bf16* __restrict__ actOut, float* __restrict__ fOut,
                            int outStride, int colOff)
{
  const int lane = threadIdx.x & 31;
  const int wv   = blockIdx.x * (blockDim.x >> 5) + (threadIdx.x >> 5);
  const int nwv  = gridDim.x * (blockDim.x >> 5);
  const int m  = lane & 15;
  const int hh = lane >> 4;
  const int K  = KC * 32;

  for (int g0 = wv; g0 < (Bn >> 4); g0 += nwv) {
    v8f c[NT];
    #pragma unroll
    for (int nt = 0; nt < NT; ++nt) {
      int col = nt*16 + m;
      float bv = 0.f;
      if (bias != nullptr && col < d) bv = bias[col];
      v8f t = {bv,bv,bv,bv,bv,bv,bv,bv};
      c[nt] = t;
    }

    const __bf16* arow = A + (size_t)(g0*16 + m) * K;
    #pragma unroll
    for (int kc = 0; kc < KC; ++kc) {
      VB af;
      af.q[0] = *(const uint4*)(arow + 32*kc + 8*hh);
      af.q[1] = *(const uint4*)(arow + 32*kc + 16 + 8*hh);
      #pragma unroll
      for (int nt = 0; nt < NT; ++nt) {
        const unsigned* bp = Wblk + ((size_t)(nt*KC + kc)*32 + lane)*8;
        VB bfr;
        bfr.q[0] = *(const uint4*)(bp);
        bfr.q[1] = *(const uint4*)(bp + 4);
        c[nt] = __builtin_amdgcn_wmma_f32_16x16x32_bf16(false, af.v, false, bfr.v, (short)0, c[nt], false, false);
      }
    }

    if (MODE == 1) {
      float s1[8] = {0,0,0,0,0,0,0,0};
      float s2[8] = {0,0,0,0,0,0,0,0};
      #pragma unroll
      for (int nt = 0; nt < NT; ++nt) {
        if (nt*16 + m < d) {
          #pragma unroll
          for (int i = 0; i < 8; ++i) { float v = c[nt][i]; s1[i] += v; s2[i] += v*v; }
        }
      }
      #pragma unroll
      for (int i = 0; i < 8; ++i) {
        #pragma unroll
        for (int off2 = 1; off2 <= 8; off2 <<= 1) {
          s1[i] += __shfl_xor(s1[i], off2, 32);
          s2[i] += __shfl_xor(s2[i], off2, 32);
        }
      }
      const float inv_d = 1.f / (float)d;
      #pragma unroll
      for (int i = 0; i < 8; ++i) {
        float mu  = s1[i] * inv_d;
        float var = s2[i] * inv_d - mu*mu;
        s1[i] = mu;
        s2[i] = __frsqrt_rn(var + 1e-5f);
      }
      #pragma unroll
      for (int nt = 0; nt < NT; ++nt) {
        int col = nt*16 + m;
        float gv = (col < d) ? gam[col]  : 0.f;
        float bt = (col < d) ? beta[col] : 0.f;
        #pragma unroll
        for (int i = 0; i < 8; ++i) {
          float v = (c[nt][i] - s1[i]) * s2[i] * gv + bt;
          c[nt][i] = v > 0.f ? v : 0.f;
        }
      }
    }

    if (MODE == 2) {
      #pragma unroll
      for (int nt = 0; nt < NT; ++nt) {
        int col = nt*16 + m;
        if (col < d) {
          #pragma unroll
          for (int i = 0; i < 8; ++i) {
            int row = g0*16 + 8*hh + i;
            fOut[(size_t)row * outStride + colOff + col] = c[nt][i];
          }
        }
      }
    } else {
      const int ldo = NT * 16;
      #pragma unroll
      for (int nt = 0; nt < NT; ++nt) {
        int col = nt*16 + m;
        #pragma unroll
        for (int i = 0; i < 8; ++i) {
          int row = g0*16 + 8*hh + i;
          actOut[(size_t)row * ldo + col] = (__bf16)c[nt][i];
        }
      }
    }
  }
}

extern "C" void kernel_launch(void* const* d_in, const int* in_sizes, int n_in,
                              void* d_out, int out_size, void* d_ws, size_t ws_size,
                              hipStream_t stream) {
  const float* emb_P   = (const float*)d_in[0];
  const float* mask_P  = (const float*)d_in[1];
  const float* emb_H   = (const float*)d_in[2];
  const float* mask_H  = (const float*)d_in[3];
  const float* pep_Bc  = (const float*)d_in[4];
  const float* pep_Ac  = (const float*)d_in[5];
  const float* pep_Hc  = (const float*)d_in[6];
  const float* pep_W1  = (const float*)d_in[7];
  const float* pep_b1  = (const float*)d_in[8];
  const float* pep_g   = (const float*)d_in[9];
  const float* pep_beta= (const float*)d_in[10];
  const float* pep_W2  = (const float*)d_in[11];
  const float* pep_b2  = (const float*)d_in[12];
  const float* hla_Bc  = (const float*)d_in[13];
  const float* hla_Ac  = (const float*)d_in[14];
  const float* hla_Hc  = (const float*)d_in[15];
  const float* hla_W1  = (const float*)d_in[16];
  const float* hla_b1  = (const float*)d_in[17];
  const float* hla_g   = (const float*)d_in[18];
  const float* hla_beta= (const float*)d_in[19];
  const float* hla_W2  = (const float*)d_in[20];
  const float* hla_b2  = (const float*)d_in[21];
  (void)n_in; (void)out_size; (void)ws_size;

  const int B = in_sizes[1] / 32;   // mask_P is [B, 32]

  char* ws = (char*)d_ws;
  size_t off = 0;
  auto take = [&](size_t bytes) -> char* {
    char* p = ws + off; off = (off + bytes + 255) & ~(size_t)255; return p;
  };
  __bf16* U_P = (__bf16*)take((size_t)B * 256 * 2);
  __bf16* U_H = (__bf16*)take((size_t)B * 256 * 2);
  __bf16* z_P = (__bf16*)take((size_t)B * 384 * 2);
  __bf16* h_P = (__bf16*)take((size_t)B * 384 * 2);
  __bf16* z_H = (__bf16*)take((size_t)B * 160 * 2);
  __bf16* h_H = (__bf16*)take((size_t)B * 160 * 2);
  unsigned* HcP = (unsigned*)take((size_t) 8 * 24 * 256 * 4);
  unsigned* W1P = (unsigned*)take((size_t)12 * 24 * 256 * 4);
  unsigned* W2P = (unsigned*)take((size_t)12 * 24 * 256 * 4);
  unsigned* HcH = (unsigned*)take((size_t) 8 * 10 * 256 * 4);
  unsigned* W1H = (unsigned*)take((size_t) 5 * 10 * 256 * 4);
  unsigned* W2H = (unsigned*)take((size_t) 5 * 10 * 256 * 4);

  auto packs = [&](const float* W, unsigned* blk, int K, int N, int KC, int NT) {
    int total = KC * NT * 256;
    pack_w_kernel<<<(total + 255) / 256, 256, 0, stream>>>(W, blk, K, N, KC, NT);
  };
  packs(pep_Hc, HcP, 256, 358,  8, 24);
  packs(pep_W1, W1P, 358, 358, 12, 24);
  packs(pep_W2, W2P, 358, 358, 12, 24);
  packs(hla_Hc, HcH, 256, 154,  8, 10);
  packs(hla_W1, W1H, 154, 154,  5, 10);
  packs(hla_W2, W2H, 154, 154,  5, 10);

  encoder_kernel<32, 1><<<512, 128, 0, stream>>>(emb_P, mask_P, pep_Bc, pep_Ac, U_P, B);
  encoder_kernel<384,12><<<512, 128, 0, stream>>>(emb_H, mask_H, hla_Bc, hla_Ac, U_H, B);

  float* out = (float*)d_out;
  gemm_kernel< 8,24,0><<<64,128,0,stream>>>(U_P, HcP, nullptr, nullptr, nullptr, 358, B, z_P, nullptr, 0, 0);
  gemm_kernel<12,24,1><<<64,128,0,stream>>>(z_P, W1P, pep_b1, pep_g, pep_beta,  358, B, h_P, nullptr, 0, 0);
  gemm_kernel<12,24,2><<<64,128,0,stream>>>(h_P, W2P, pep_b2, nullptr, nullptr, 358, B, nullptr, out, 512, 0);
  gemm_kernel< 8,10,0><<<64,128,0,stream>>>(U_H, HcH, nullptr, nullptr, nullptr, 154, B, z_H, nullptr, 0, 0);
  gemm_kernel< 5,10,1><<<64,128,0,stream>>>(z_H, W1H, hla_b1, hla_g, hla_beta,  154, B, h_H, nullptr, 0, 0);
  gemm_kernel< 5,10,2><<<64,128,0,stream>>>(h_H, W2H, hla_b2, nullptr, nullptr, 154, B, nullptr, out, 512, 358);
}
